// TaggerBiRNN_79817672228902
// MI455X (gfx1250) — matrix operand order repaired and checked by reading, not compiled, not run on hardware
//
#include <hip/hip_runtime.h>
#include <hip/hip_bf16.h>

// ---------------- problem constants ----------------
#define VOCAB 50000
#define EMB   300
#define HID   100
#define NCLS  20
#define BATCH 128
#define SEQ   512
#define BS    (BATCH * SEQ)

// padded dims for WMMA tiling
#define EPAD  320          // EMB padded to multiple of 32 (K of big GEMM)
#define NPAD  320          // 3*HID padded (rows of packed weights)
#define KHP   128          // HID padded to multiple of 32 (K of recurrence GEMM)
#define GIP   304          // row stride (floats) of GI intermediate

typedef __attribute__((ext_vector_type(16))) _Float16 v16h;
typedef __attribute__((ext_vector_type(8)))  _Float16 v8h;
typedef __attribute__((ext_vector_type(8)))  float    v8f;
typedef __attribute__((ext_vector_type(4)))  unsigned u32x4;
typedef __attribute__((ext_vector_type(8)))  int      i32x8;
typedef __attribute__((ext_vector_type(4)))  int      i32x4;

__device__ __forceinline__ unsigned hash_u32(unsigned x) {
  x ^= x >> 16; x *= 0x7feb352dU;
  x ^= x >> 15; x *= 0x846ca68bU;
  x ^= x >> 16;
  return x;
}

// ---------------------------------------------------------------------------
// Kernel 0: pack w_ih [3H,EMB] -> f16 [NPAD][EPAD], w_hh [3H,HID] -> f16 [NPAD][KHP]
// "row = output unit n, contiguous K" order so WMMA B-fragments (32x16, lane =
// column n, 16 contiguous K per lane) are single 32B loads.
// ---------------------------------------------------------------------------
__global__ void pack_weights_k(const float* __restrict__ w_ih,
                               const float* __restrict__ w_hh,
                               _Float16* __restrict__ Wih,
                               _Float16* __restrict__ Whh) {
  int idx = blockIdx.x * blockDim.x + threadIdx.x;
  const int NW1 = NPAD * EPAD;            // 102400
  const int NW2 = NPAD * KHP;             // 40960
  if (idx < NW1) {
    int n = idx / EPAD, k = idx % EPAD;
    float v = (n < 3 * HID && k < EMB) ? w_ih[n * EMB + k] : 0.0f;
    Wih[idx] = (_Float16)v;
  } else if (idx < NW1 + NW2) {
    int r = idx - NW1;
    int n = r / KHP, k = r % KHP;
    float v = (n < 3 * HID && k < HID) ? w_hh[n * HID + k] : 0.0f;
    Whh[r] = (_Float16)v;
  }
}

// ---------------------------------------------------------------------------
// Kernel 1: embedding gather + dropout1 + f32->f16 cast into Xh [BS][EPAD]
// ---------------------------------------------------------------------------
__global__ void embed_drop_k(const int* __restrict__ tok,
                             const float* __restrict__ emb,
                             _Float16* __restrict__ Xh) {
  const int row = blockIdx.x;                 // token index b*SEQ + s
  const int t = tok[row];
  const float* e = emb + (size_t)t * EMB;
  for (int c = threadIdx.x; c < EPAD; c += blockDim.x) {
    float v = 0.0f;
    if (c < EMB) {
      v = e[c];
      unsigned h = hash_u32(0x85EBCA6BU ^ ((unsigned)row * (unsigned)EMB + (unsigned)c));
      v = (h & 1U) ? v * 2.0f : 0.0f;         // keep=0.5 -> scale 1/keep = 2
    }
    Xh[(size_t)row * EPAD + c] = (_Float16)v;
  }
}

// ---------------------------------------------------------------------------
// Kernel 2: GI[BS,300] = Xh[BS,EPAD] @ Wih^T + b_ih   (WMMA f16 -> f32)
// Block = 128 threads = 4 waves; each wave owns a 16(M) x 64(N) output tile
// with 4 accumulators, so each A fragment is loaded once per K-step and
// reused across 4 WMMAs (4x less L2 A-traffic). Grid = (BS/64, 5 N-groups).
// ---------------------------------------------------------------------------
__global__ void gi_gemm_k(const _Float16* __restrict__ Xh,
                          const _Float16* __restrict__ Wih,
                          const float* __restrict__ b_ih,
                          float* __restrict__ GI) {
  const int lane  = threadIdx.x & 31;
  const int wave  = threadIdx.x >> 5;
  const int m0    = (blockIdx.x * 4 + wave) * 16;
  const int n0    = blockIdx.y * 64;
  const int mrow  = m0 + (lane & 15);
  const int khalf = lane >> 4;                 // which 16-lane half
  const int nb    = n0 + (lane & 15);

  const _Float16* arow = Xh + (size_t)mrow * EPAD;
  const _Float16* bp0 = Wih + (size_t)(nb     ) * EPAD + khalf * 16;
  const _Float16* bp1 = Wih + (size_t)(nb + 16) * EPAD + khalf * 16;
  const _Float16* bp2 = Wih + (size_t)(nb + 32) * EPAD + khalf * 16;
  const _Float16* bp3 = Wih + (size_t)(nb + 48) * EPAD + khalf * 16;
  __builtin_prefetch(bp0, 0, 0);               // speculative L2 prefetch

  v8f acc0 = {0.f,0.f,0.f,0.f,0.f,0.f,0.f,0.f};
  v8f acc1 = acc0, acc2 = acc0, acc3 = acc0;
#pragma unroll
  for (int kb = 0; kb < EPAD / 32; ++kb) {
    const int k0 = kb * 32;
    union { v16h v; v8h h[2]; } a;
    // A 16x32 f16 fragment: lane half selects K sub-block (ISA 7.12.2)
    a.h[0] = *(const v8h*)(arow + k0 + khalf * 16);       // K = k0+16*khalf+{0..7}
    a.h[1] = *(const v8h*)(arow + k0 + khalf * 16 + 8);   // K = k0+16*khalf+{8..15}
    // B 32x16 f16 fragments: 16 contiguous K per lane; A reused 4x
    v16h b;
    b = *(const v16h*)(bp0 + k0);
    acc0 = __builtin_amdgcn_wmma_f32_16x16x32_f16(false, a.v, false, b, (short)0, acc0, false, false);
    b = *(const v16h*)(bp1 + k0);
    acc1 = __builtin_amdgcn_wmma_f32_16x16x32_f16(false, a.v, false, b, (short)0, acc1, false, false);
    b = *(const v16h*)(bp2 + k0);
    acc2 = __builtin_amdgcn_wmma_f32_16x16x32_f16(false, a.v, false, b, (short)0, acc2, false, false);
    b = *(const v16h*)(bp3 + k0);
    acc3 = __builtin_amdgcn_wmma_f32_16x16x32_f16(false, a.v, false, b, (short)0, acc3, false, false);
  }
  v8f accs[4] = {acc0, acc1, acc2, acc3};
#pragma unroll
  for (int u = 0; u < 4; ++u) {
    const int nn = nb + u * 16;
    if (nn < 3 * HID) {
      const float bias = b_ih[nn];
#pragma unroll
      for (int j = 0; j < 8; ++j) {
        const int m = m0 + j + 8 * (lane >> 4);  // C layout: VGPR j -> row j / j+8
        GI[(size_t)m * GIP + nn] = accs[u][j] + bias;
      }
    }
  }
}

// ---------------------------------------------------------------------------
// Kernel 3: GRU recurrence + dropout2 + classifier + log_softmax.
// 8 workgroups x 256 threads; each WG owns 16 batch rows and loops t=0..511.
// h state lives in LDS; w_hh (80 KB) is staged once into dynamic LDS via the
// Tensor Data Mover (fallback: vectorized copy), serving all 512 steps'
// B-fragments from ds_load_b128 instead of 512 global re-reads.
// ---------------------------------------------------------------------------
__global__ void gru_rnn_k(const float* __restrict__ GI,
                          const _Float16* __restrict__ Whh,
                          const float* __restrict__ b_hh,
                          const float* __restrict__ w_lin,
                          const float* __restrict__ b_lin,
                          float* __restrict__ out) {
  extern __shared__ char dynlds[];
  _Float16* hWhh = (_Float16*)dynlds;      // [NPAD * KHP] f16 = 81920 B

  __shared__ _Float16 hA[16 * KHP];        // f16 h, WMMA-A source (zero padded K)
  __shared__ float    hF[16 * HID];        // f32 h state
  __shared__ float    gh[16 * GIP];        // h @ w_hh^T result
  __shared__ float    hd[16 * HID];        // dropped h for classifier
  __shared__ float    logits[16 * NCLS];
  __shared__ float    rowMax[16], rowLse[16];
  __shared__ float    bhh_s[3 * HID];
  __shared__ float    wlin_s[NCLS * HID];
  __shared__ float    blin_s[NCLS];

  const int tid  = threadIdx.x;
  const int lane = tid & 31;
  const int wave = tid >> 5;
  const int b0   = blockIdx.x * 16;

  // ---- stage w_hh into LDS ----
#if __has_builtin(__builtin_amdgcn_tensor_load_to_lds) && __has_builtin(__builtin_amdgcn_s_wait_tensorcnt)
  if (wave == 0) {
    // Tensor DMA descriptor (CDNA5 ISA): 2B elements,
    // one 40960-element row, global Whh -> LDS hWhh.
    const unsigned ldsb = (unsigned)(size_t)(void*)hWhh;     // LDS aperture: addr[31:0] = LDS offset
    const unsigned long long ga = (unsigned long long)(size_t)Whh;
    u32x4 g0;
    g0[0] = 1u;                                              // count=1 (valid descriptor)
    g0[1] = ldsb;                                            // lds_addr
    g0[2] = (unsigned)(ga & 0xFFFFFFFFu);                    // global_addr[31:0]
    g0[3] = (unsigned)((ga >> 32) & 0x01FFFFFFu) | (2u << 30); // global_addr[56:32] | type=2
    i32x8 g1;
    g1[0] = (int)(1u << 16);                                 // data_size=1 (2 bytes), no mask
    g1[1] = (int)((40960u & 0xFFFFu) << 16);                 // tensor_dim0[15:0] in bits 63:48
    g1[2] = (int)(((40960u >> 16) & 0xFFFFu) | (1u << 16));  // tensor_dim0 hi | tensor_dim1=1
    g1[3] = (int)((40960u & 0xFFFFu) << 16);                 // tile_dim0=40960 in bits 127:112
    g1[4] = 1;                                               // tile_dim1=1
    g1[5] = 40960;                                           // tensor_dim0_stride[31:0]
    g1[6] = 0;
    g1[7] = 0;
    i32x4 g2 = {0, 0, 0, 0};
    i32x4 g3 = {0, 0, 0, 0};
    i32x8 g4 = {0, 0, 0, 0, 0, 0, 0, 0};                     // 6-arg form: extra group, zero
    __builtin_amdgcn_tensor_load_to_lds(g0, g1, g2, g3, g4, 0);
    __builtin_amdgcn_s_wait_tensorcnt(0);
  }
#else
  for (int i = tid; i < (NPAD * KHP) / 8; i += 256)
    ((v8h*)hWhh)[i] = ((const v8h*)Whh)[i];
#endif

  for (int i = tid; i < 16 * KHP; i += 256) hA[i] = (_Float16)0.0f;
  for (int i = tid; i < 16 * HID; i += 256) hF[i] = 0.0f;
  for (int i = tid; i < 3 * HID; i += 256)  bhh_s[i] = b_hh[i];
  for (int i = tid; i < NCLS * HID; i += 256) wlin_s[i] = w_lin[i];
  if (tid < NCLS) blin_s[tid] = b_lin[tid];
  __syncthreads();

  for (int t = 0; t < SEQ; ++t) {
    // ---- phase 1: gh = h @ w_hh^T via WMMA (A and B both from LDS) ----
    {
      const int khalf = lane >> 4;
      const int mloc  = lane & 15;
      const _Float16* ap = &hA[mloc * KHP] + khalf * 16;
      for (int nt = wave; nt < 19; nt += 8) {
        const int nn = nt * 16 + (lane & 15);
        const _Float16* bp = &hWhh[nn * KHP] + khalf * 16;
        v8f c = {0.f, 0.f, 0.f, 0.f, 0.f, 0.f, 0.f, 0.f};
#pragma unroll
        for (int kb = 0; kb < KHP / 32; ++kb) {
          const int k0 = kb * 32;
          union { v16h v; v8h h[2]; } a, b;
          a.h[0] = *(const v8h*)(ap + k0);        // ds_load_b128
          a.h[1] = *(const v8h*)(ap + k0 + 8);
          b.v = *(const v16h*)(bp + k0);
          c = __builtin_amdgcn_wmma_f32_16x16x32_f16(false, a.v, false, b.v,
                                                     (short)0, c, false, false);
        }
        if (nn < 3 * HID) {
#pragma unroll
          for (int j = 0; j < 8; ++j) {
            const int m = j + 8 * (lane >> 4);
            gh[m * GIP + nn] = c[j];
          }
        }
      }
    }
    __syncthreads();

    // ---- phase 2: gates, h update, dropout2 (each (m,j) owned by one thread) ----
    for (int idx = tid; idx < 16 * HID; idx += 256) {
      const int m = idx / HID, j = idx % HID;
      const size_t row = (size_t)(b0 + m) * SEQ + t;
      const float* gi = GI + row * GIP;
      const float ir = gi[j], iz = gi[HID + j], inn = gi[2 * HID + j];
      const float hr = gh[m * GIP + j]           + bhh_s[j];
      const float hz = gh[m * GIP + HID + j]     + bhh_s[HID + j];
      const float hn = gh[m * GIP + 2 * HID + j] + bhh_s[2 * HID + j];
      const float r  = 1.0f / (1.0f + __expf(-(ir + hr)));
      const float z  = 1.0f / (1.0f + __expf(-(iz + hz)));
      const float n  = tanhf(inn + r * hn);
      const float hp = hF[m * HID + j];
      const float hnew = (1.0f - z) * n + z * hp;
      hF[m * HID + j] = hnew;
      hA[m * KHP + j] = (_Float16)hnew;
      const unsigned h = hash_u32(0x9E3779B9U ^
          (((unsigned)t * (unsigned)BATCH + (unsigned)(b0 + m)) * (unsigned)HID + (unsigned)j));
      hd[m * HID + j] = (h & 1U) ? hnew * 2.0f : 0.0f;
    }
    __syncthreads();

    // ---- phase 3: classifier logits ----
    for (int idx = tid; idx < 16 * NCLS; idx += 256) {
      const int m = idx / NCLS, c = idx % NCLS;
      float acc = blin_s[c];
      for (int j = 0; j < HID; ++j) acc += hd[m * HID + j] * wlin_s[c * HID + j];
      logits[m * NCLS + c] = acc;
    }
    __syncthreads();

    // ---- phase 4: per-row max / logsumexp ----
    if (tid < 16) {
      float mx = -1e30f;
      for (int c = 0; c < NCLS; ++c) mx = fmaxf(mx, logits[tid * NCLS + c]);
      float s = 0.0f;
      for (int c = 0; c < NCLS; ++c) s += __expf(logits[tid * NCLS + c] - mx);
      rowMax[tid] = mx;
      rowLse[tid] = __logf(s);
    }
    __syncthreads();

    // ---- phase 5: write y[b, c, t] ----
    for (int idx = tid; idx < 16 * NCLS; idx += 256) {
      const int m = idx / NCLS, c = idx % NCLS;
      out[(size_t)(b0 + m) * NCLS * SEQ + (size_t)c * SEQ + t] =
          logits[m * NCLS + c] - rowMax[m] - rowLse[m];
    }
    __syncthreads();
  }
}

// ---------------------------------------------------------------------------
extern "C" void kernel_launch(void* const* d_in, const int* in_sizes, int n_in,
                              void* d_out, int out_size, void* d_ws, size_t ws_size,
                              hipStream_t stream) {
  (void)in_sizes; (void)n_in; (void)out_size; (void)ws_size;
  const int*   tok   = (const int*)d_in[0];
  const float* emb   = (const float*)d_in[1];
  const float* w_ih  = (const float*)d_in[2];
  const float* w_hh  = (const float*)d_in[3];
  const float* b_ih  = (const float*)d_in[4];
  const float* b_hh  = (const float*)d_in[5];
  const float* w_lin = (const float*)d_in[6];
  const float* b_lin = (const float*)d_in[7];
  float* out = (float*)d_out;

  // workspace layout (all offsets 256B aligned)
  char* ws = (char*)d_ws;
  _Float16* Wih = (_Float16*)(ws + 0);                        // 320*320*2   = 204800
  _Float16* Whh = (_Float16*)(ws + 204800);                   // 320*128*2   =  81920
  _Float16* Xh  = (_Float16*)(ws + 286720);                   // BS*320*2    = 41943040
  float*    GI  = (float*)   (ws + 42229760);                 // BS*304*4    = 79691776

  pack_weights_k<<<(NPAD * EPAD + NPAD * KHP + 255) / 256, 256, 0, stream>>>(
      w_ih, w_hh, Wih, Whh);
  embed_drop_k<<<BS, 256, 0, stream>>>(tok, emb, Xh);
  gi_gemm_k<<<dim3(BS / 64, 5), 128, 0, stream>>>(Xh, Wih, b_ih, GI);
  gru_rnn_k<<<BATCH / 16, 256, (size_t)(NPAD * KHP * 2), stream>>>(
      GI, Whh, b_hh, w_lin, b_lin, out);
}
